// ResonatorBlock_12996571038029
// MI455X (gfx1250) — matrix operand order repaired and checked by reading, not compiled, not run on hardware
//
#include <hip/hip_runtime.h>
#include <math.h>

#define DIMN  4096
#define BATCH 4096
#define DT_C  0.1f
#define PI_F      3.14159265358979323846f
#define TWO_PI_F  6.28318530717958647692f

// LDS staging geometry for the kappa tile: 64 rows x 32 halves per K-step.
#define KROWS      64
#define KROW_BYTES 64            // 32 halves of payload per row
#define KROW_PITCH 80            // padded pitch (multiple of 16B, spreads banks)

typedef __attribute__((ext_vector_type(16))) _Float16 v16h;
typedef __attribute__((ext_vector_type(8)))  float    v8f;

// ---------------------------------------------------------------------------
// Pass 1: elementwise precompute.
//   S[idx] = (f16) sin(x[idx]),  C[idx] = (f16) cos(x[idx]),  Kh = (f16) kappa
// ---------------------------------------------------------------------------
__global__ __launch_bounds__(256) void resonator_precompute(
    const float* __restrict__ x, const float* __restrict__ kappa,
    _Float16* __restrict__ S, _Float16* __restrict__ C,
    _Float16* __restrict__ Kh, int n)
{
    int idx = blockIdx.x * blockDim.x + threadIdx.x;
    if (idx < n) {
        float v = x[idx];
        float s, c;
        __sincosf(v, &s, &c);
        S[idx]  = (_Float16)s;
        C[idx]  = (_Float16)c;
        Kh[idx] = (_Float16)kappa[idx];
    }
}

// ---------------------------------------------------------------------------
// Fragment loader: two 16-byte chunks -> one 16-half WMMA operand.
// ---------------------------------------------------------------------------
static __device__ __forceinline__ v16h load_frag2(const _Float16* p0,
                                                  const _Float16* p1)
{
    union { uint4 u[2]; v16h h; } cvt;
    cvt.u[0] = *reinterpret_cast<const uint4*>(p0);
    cvt.u[1] = *reinterpret_cast<const uint4*>(p1);
    return cvt.h;
}

// ---------------------------------------------------------------------------
// Pass 2: dual GEMM (kc = C·kappa^T, ks = S·kappa^T) fused with epilogue.
// Block = 256 threads = 8 waves. Block tile: 128 rows (batch) x 64 cols.
// kappa tile staged in LDS via async-to-LDS loads, double buffered.
// ---------------------------------------------------------------------------
__global__ __launch_bounds__(256) void resonator_wmma_gemm(
    const float* __restrict__ x, const float* __restrict__ omega,
    const _Float16* __restrict__ S, const _Float16* __restrict__ C,
    const _Float16* __restrict__ Kh, float* __restrict__ out)
{
    __shared__ __align__(16) unsigned char kbuf[2][KROWS * KROW_PITCH];

    const int tid   = threadIdx.x;
    const int lane  = tid & 31;
    const int wave  = tid >> 5;
    const int lrow  = lane & 15;   // M (A frag) / N (B frag, C/D frag)
    const int lhi   = lane >> 4;   // which half of the wave

    const int Mblock = blockIdx.y * 128;
    const int Nblock = blockIdx.x * 64;
    const int Mw     = Mblock + wave * 16;

    // A-fragment base rows (sin / cos of this wave's 16 batch rows)
    const _Float16* srow = S + (size_t)(Mw + lrow) * DIMN;
    const _Float16* crow = C + (size_t)(Mw + lrow) * DIMN;

    // Async staging assignment: thread -> (kappa row, 16B segment)
    const int srow_idx = tid >> 2;                 // 0..63
    const int sseg     = tid & 3;                  // 0..3 (16B each)
    const _Float16* gsrc_base =
        Kh + (size_t)(Nblock + srow_idx) * DIMN + sseg * 8;
    const int lds_row_off = srow_idx * KROW_PITCH + sseg * 16;

    v8f acc_c[4];  // kc accumulators (A = cos)
    v8f acc_s[4];  // ks accumulators (A = sin)
#pragma unroll
    for (int t = 0; t < 4; ++t) { acc_c[t] = (v8f)(0.0f); acc_s[t] = (v8f)(0.0f); }

    const int aoff0 = lhi * 8;        // A frag halves 0..7  : K = lhi*8 ..
    const int aoff1 = 16 + lhi * 8;   // A frag halves 8..15 : K = 16+lhi*8 ..

    // Prologue: stage K-step 0 into buffer 0.
    {
        unsigned lds_addr =
            (unsigned)(size_t)(&kbuf[0][0] + lds_row_off);
        unsigned long long ga = (unsigned long long)(size_t)(gsrc_base + 0);
        asm volatile("global_load_async_to_lds_b128 %0, %1, off"
                     :: "v"(lds_addr), "v"(ga) : "memory");
    }

    int cur = 0;
    for (int kb = 0; kb < DIMN; kb += 32) {
        // My async writes into kbuf[cur] have landed in LDS.
        asm volatile("s_wait_asynccnt 0x0" ::: "memory");
        // Everyone's writes landed; everyone's reads of kbuf[cur^1] from the
        // previous iteration were consumed (WMMAs precede this barrier).
        __syncthreads();

        // Kick off the next K-step's kappa tile into the other buffer.
        if (kb + 32 < DIMN) {
            unsigned lds_addr =
                (unsigned)(size_t)(&kbuf[cur ^ 1][0] + lds_row_off);
            unsigned long long ga =
                (unsigned long long)(size_t)(gsrc_base + kb + 32);
            asm volatile("global_load_async_to_lds_b128 %0, %1, off"
                         :: "v"(lds_addr), "v"(ga) : "memory");
        }

        // Prefetch next A-stream cache lines.
        if (kb + 32 < DIMN) {
            __builtin_prefetch(srow + kb + 32 + aoff0, 0, 1);
            __builtin_prefetch(crow + kb + 32 + aoff0, 0, 1);
        }

        v16h a_s = load_frag2(srow + kb + aoff0, srow + kb + aoff1);
        v16h a_c = load_frag2(crow + kb + aoff0, crow + kb + aoff1);

        const unsigned char* bbase = &kbuf[cur][0];
#pragma unroll
        for (int t = 0; t < 4; ++t) {
            // B frag: lane = column N, halves 0..7: K = lhi*8.., halves 8..15: K = 16+lhi*8..
            const _Float16* bp = (const _Float16*)
                (bbase + (t * 16 + lrow) * KROW_PITCH + lhi * 16);
            v16h b = load_frag2(bp, bp + 16);
            acc_c[t] = __builtin_amdgcn_wmma_f32_16x16x32_f16(
                false, a_c, false, b, (short)0, acc_c[t], false, false);
            acc_s[t] = __builtin_amdgcn_wmma_f32_16x16x32_f16(
                false, a_s, false, b, (short)0, acc_s[t], false, false);
        }
        cur ^= 1;
    }

    // Epilogue: D-matrix layout -> lane = N (lrow), vgpr r -> M = r + 8*lhi.
#pragma unroll
    for (int t = 0; t < 4; ++t) {
        const int col = Nblock + t * 16 + lrow;
        const float om = omega[col];
#pragma unroll
        for (int r = 0; r < 8; ++r) {
            const int row = Mw + r + 8 * lhi;
            const float kc = acc_c[t][r];
            const float ks = acc_s[t][r];
            const float xv = x[(size_t)row * DIMN + col];
            float s, c;
            __sincosf(xv, &s, &c);
            const float u  = s * kc - c * ks;
            const float nx = xv + DT_C * (om + u);
            float m = fmodf(nx + PI_F, TWO_PI_F);
            if (m < 0.0f) m += TWO_PI_F;
            out[(size_t)row * DIMN + col] = m - PI_F;
        }
    }
}

// ---------------------------------------------------------------------------
// Launch
// ---------------------------------------------------------------------------
extern "C" void kernel_launch(void* const* d_in, const int* in_sizes, int n_in,
                              void* d_out, int out_size, void* d_ws, size_t ws_size,
                              hipStream_t stream)
{
    const float* x     = (const float*)d_in[0];   // [BATCH, DIMN]
    const float* kappa = (const float*)d_in[1];   // [DIMN, DIMN]
    const float* omega = (const float*)d_in[2];   // [DIMN]
    float* out = (float*)d_out;

    const size_t nn = (size_t)DIMN * DIMN;
    _Float16* S  = (_Float16*)d_ws;   // 32 MB
    _Float16* C  = S + nn;            // 32 MB
    _Float16* Kh = C + nn;            // 32 MB  (total 96 MB of ws)

    const int n = (int)nn;
    resonator_precompute<<<n / 256, 256, 0, stream>>>(x, kappa, S, C, Kh, n);

    dim3 grid(DIMN / 64, BATCH / 128);  // (N blocks, M blocks)
    resonator_wmma_gemm<<<grid, 256, 0, stream>>>(x, omega, S, C, Kh, out);
}
